// Attention_4887672783355
// MI455X (gfx1250) — matrix operand order repaired and checked by reading, not compiled, not run on hardware
//
#include <hip/hip_runtime.h>

// ---------------------------------------------------------------------------
// BERT self-attention block for MI455X (gfx1250, wave32, WMMA).
//  B=2, S=4096, H=768, NH=12, HD=64.
//  Pipeline: f32->f16 convert -> QKV WMMA GEMM -> flash attention (WMMA,
//            async K-tile staging to LDS) -> O-proj WMMA GEMM + bias +
//            residual -> in-place LayerNorm.
//  Workspace use: ~65 MB of d_ws for f16 staging buffers.
// ---------------------------------------------------------------------------

typedef _Float16 half_t;
typedef __attribute__((ext_vector_type(16))) _Float16 v16h;
typedef __attribute__((ext_vector_type(8)))  _Float16 v8h;
typedef __attribute__((ext_vector_type(8)))  float    v8f;

#define BD    2
#define SD    4096
#define HDIM  768
#define NHD   12
#define HDD   64
#define MROWS (BD * SD)   // 8192

__device__ __forceinline__ v16h pack16(v8h lo, v8h hi) {
  v16h f;
#pragma unroll
  for (int i = 0; i < 8; i++) { f[i] = lo[i]; f[i + 8] = hi[i]; }
  return f;
}

// A fragment (16 rows x 32 K), row-major source, leading dim ld.
// 16-bit A layout: lanes 0-15 -> rows 0-15, K = {0..7, 16..23};
//                  lanes 16-31 -> rows 0-15, K = {8..15, 24..31}.
__device__ __forceinline__ v16h load_fragA(const half_t* base, int ld, int row0, int k0) {
  const int lane = threadIdx.x & 31;
  const half_t* p = base + (size_t)(row0 + (lane & 15)) * ld + k0 + ((lane >> 4) << 3);
  v8h lo = *(const v8h*)p;
  v8h hi = *(const v8h*)(p + 16);
  return pack16(lo, hi);
}

// B fragment (32 K x 16 cols) where each COLUMN is contiguous along K in the
// source with leading dim ld (true for W rows in X @ W^T, K^T tiles, V^T tiles).
// 16-bit B layout: lanes 0-15 -> cols 0-15, K = {0..7, 16..23};
//                  lanes 16-31 -> cols 0-15, K = {8..15, 24..31}.
__device__ __forceinline__ v16h load_fragB(const half_t* base, int ld, int col0, int k0) {
  const int lane = threadIdx.x & 31;
  const half_t* p = base + (size_t)(col0 + (lane & 15)) * ld + k0 + ((lane >> 4) << 3);
  v8h lo = *(const v8h*)p;
  v8h hi = *(const v8h*)(p + 16);
  return pack16(lo, hi);
}

__device__ __forceinline__ v8f wmma_ff(v16h a, v16h b, v8f c) {
  // D = A(16x32 f16) * B(32x16 f16) + C(16x16 f32)
  return __builtin_amdgcn_wmma_f32_16x16x32_f16(false, a, false, b, (short)0, c,
                                                false, false);
}

// CDNA5 async global->LDS copy (16B per lane), tracked by ASYNCcnt.
// Operands: vdst = LDS byte address (32-bit VGPR), vaddr = 64-bit global addr.
__device__ __forceinline__ void async_load_b128(unsigned lds_addr, const void* gsrc) {
  asm volatile("global_load_async_to_lds_b128 %0, %1, off"
               :: "v"(lds_addr), "v"((unsigned long long)(size_t)gsrc)
               : "memory");
}

__device__ __forceinline__ void wait_async0() {
#if __has_builtin(__builtin_amdgcn_s_wait_asynccnt)
  __builtin_amdgcn_s_wait_asynccnt(0);
#else
  asm volatile("s_wait_asynccnt 0x0" ::: "memory");
#endif
}

// ---------------------------------------------------------------------------
__global__ __launch_bounds__(256) void convert_kernel(const float* __restrict__ in,
                                                      half_t* __restrict__ out, int n) {
  int i = blockIdx.x * 256 + threadIdx.x;
  if (i < n) out[i] = (half_t)in[i];
}

// ---------------------------------------------------------------------------
// Out[m,n] = sum_k A[m,k] * W[n,k] + bias[n]   (f16 out)
// grid = (HDIM/128, MROWS/128), block = 256 (8 waves, 4x2 wave tiling, 32x64/wave)
__global__ __launch_bounds__(256) void gemm_qkv(const half_t* __restrict__ A,
                                                const half_t* __restrict__ W,
                                                const float* __restrict__ bias,
                                                half_t* __restrict__ Out) {
  const int lane = threadIdx.x & 31;
  const int wave = threadIdx.x >> 5;
  const int m0 = blockIdx.y * 128 + (wave >> 1) * 32;
  const int n0 = blockIdx.x * 128 + (wave & 1) * 64;

  v8f acc[2][4] = {};
  for (int k0 = 0; k0 < HDIM; k0 += 32) {
    v16h a0 = load_fragA(A, HDIM, m0, k0);
    v16h a1 = load_fragA(A, HDIM, m0 + 16, k0);
#pragma unroll
    for (int j = 0; j < 4; j++) {
      v16h b = load_fragB(W, HDIM, n0 + 16 * j, k0);
      acc[0][j] = wmma_ff(a0, b, acc[0][j]);
      acc[1][j] = wmma_ff(a1, b, acc[1][j]);
    }
  }

  const int col = lane & 15;
  const int rbase = (lane >> 4) << 3;
#pragma unroll
  for (int i = 0; i < 2; i++)
#pragma unroll
    for (int j = 0; j < 4; j++) {
      int n = n0 + 16 * j + col;
      float bv = bias[n];
#pragma unroll
      for (int r = 0; r < 8; r++) {
        int m = m0 + 16 * i + rbase + r;
        Out[(size_t)m * HDIM + n] = (half_t)(acc[i][j][r] + bv);
      }
    }
}

// Out[m,n] = sum_k A[m,k] * W[n,k] + bias[n] + resid[m,n]   (f32 out)
__global__ __launch_bounds__(256) void gemm_oproj(const half_t* __restrict__ A,
                                                  const half_t* __restrict__ W,
                                                  const float* __restrict__ bias,
                                                  const float* __restrict__ resid,
                                                  float* __restrict__ Out) {
  const int lane = threadIdx.x & 31;
  const int wave = threadIdx.x >> 5;
  const int m0 = blockIdx.y * 128 + (wave >> 1) * 32;
  const int n0 = blockIdx.x * 128 + (wave & 1) * 64;

  v8f acc[2][4] = {};
  for (int k0 = 0; k0 < HDIM; k0 += 32) {
    v16h a0 = load_fragA(A, HDIM, m0, k0);
    v16h a1 = load_fragA(A, HDIM, m0 + 16, k0);
#pragma unroll
    for (int j = 0; j < 4; j++) {
      v16h b = load_fragB(W, HDIM, n0 + 16 * j, k0);
      acc[0][j] = wmma_ff(a0, b, acc[0][j]);
      acc[1][j] = wmma_ff(a1, b, acc[1][j]);
    }
  }

  const int col = lane & 15;
  const int rbase = (lane >> 4) << 3;
#pragma unroll
  for (int i = 0; i < 2; i++)
#pragma unroll
    for (int j = 0; j < 4; j++) {
      int n = n0 + 16 * j + col;
      float bv = bias[n];
#pragma unroll
      for (int r = 0; r < 8; r++) {
        size_t idx = (size_t)(m0 + 16 * i + rbase + r) * HDIM + n;
        Out[idx] = acc[i][j][r] + bv + resid[idx];
      }
    }
}

// ---------------------------------------------------------------------------
// Flash attention: grid = (S/128, NH, B), block = 256 (8 waves).
// Each wave owns 16 query rows; 64-key tiles staged in LDS (K via async
// global->LDS DMA, V transposed through VGPRs); online softmax.
__global__ __launch_bounds__(256) void attn_kernel(const half_t* __restrict__ Q,
                                                   const half_t* __restrict__ K,
                                                   const half_t* __restrict__ V,
                                                   const float* __restrict__ mask,
                                                   half_t* __restrict__ Ctx) {
  __shared__ half_t Kt[64 * 64];       // [key][d] row-major (async-filled)
  __shared__ half_t Vt[64 * 72];       // [d][key] transposed (padded stride 72)
  __shared__ float  smask[64];
  __shared__ half_t Pt[8][16 * 64];    // per-wave probability tile

  const int tid = threadIdx.x;
  const int lane = tid & 31;
  const int wave = tid >> 5;
  const int b = blockIdx.z;
  const int h = blockIdx.y;
  const int q0 = blockIdx.x * 128;
  const size_t headoff = (size_t)b * SD * HDIM + (size_t)h * HDD;
  const half_t* Qb = Q + headoff;
  const half_t* Kb = K + headoff;
  const half_t* Vb = V + headoff;
  const int qrow = q0 + wave * 16;
  const int col = lane & 15;
  const int half_id = lane >> 4;

  // Q fragments, pre-scaled by 1/sqrt(HD) = 0.125
  v16h qa[2];
#pragma unroll
  for (int c = 0; c < 2; c++) {
    v16h t = load_fragA(Qb, HDIM, qrow, 32 * c);
#pragma unroll
    for (int i = 0; i < 16; i++) t[i] = (half_t)((float)t[i] * 0.125f);
    qa[c] = t;
  }

  v8f acc[4] = {};
  float mrow[8], lrow[8];
#pragma unroll
  for (int r = 0; r < 8; r++) { mrow[r] = -1e30f; lrow[r] = 0.f; }

  for (int kb = 0; kb < SD; kb += 64) {
    // --- tile stage: K async -> LDS, V via VGPR transpose, mask --------------
    {
      const int key = tid >> 2;           // 0..63
      const int dc = (tid & 3) << 4;      // 0,16,32,48
      const half_t* ks = Kb + (size_t)(kb + key) * HDIM + dc;
      async_load_b128((unsigned)(size_t)&Kt[key * 64 + dc], ks);
      async_load_b128((unsigned)(size_t)&Kt[key * 64 + dc + 8], ks + 8);

      const half_t* vs = Vb + (size_t)(kb + key) * HDIM + dc;
      v8h v0v = *(const v8h*)vs;
      v8h v1v = *(const v8h*)(vs + 8);
#pragma unroll
      for (int i = 0; i < 8; i++) {
        Vt[(dc + i) * 72 + key] = v0v[i];
        Vt[(dc + 8 + i) * 72 + key] = v1v[i];
      }
      if (tid < 64) smask[tid] = mask[(size_t)b * SD + kb + tid];
      if (kb + 64 < SD) {
        __builtin_prefetch((const void*)(vs + (size_t)64 * HDIM), 0, 1);
      }
      wait_async0();  // K tile resident in LDS
    }
    __syncthreads();

    // --- scores: S = (Q/8) @ K^T + mask -------------------------------------
    v8f sc4[4];
#pragma unroll
    for (int j = 0; j < 4; j++) {
      v8f s = {};
      s = wmma_ff(qa[0], load_fragB(Kt, 64, 16 * j, 0), s);
      s = wmma_ff(qa[1], load_fragB(Kt, 64, 16 * j, 32), s);
      float mk = smask[16 * j + col];
#pragma unroll
      for (int r = 0; r < 8; r++) s[r] += mk;
      sc4[j] = s;
    }

    // --- online softmax (row = r + 8*half_id, cols across 16-lane group) ----
    float scale[8], rsum[8];
#pragma unroll
    for (int r = 0; r < 8; r++) {
      float v = fmaxf(fmaxf(sc4[0][r], sc4[1][r]), fmaxf(sc4[2][r], sc4[3][r]));
#pragma unroll
      for (int off = 1; off < 16; off <<= 1) v = fmaxf(v, __shfl_xor(v, off, 32));
      float mold = mrow[r];
      float mn = fmaxf(mold, v);
      scale[r] = expf(mold - mn);
      mrow[r] = mn;
      rsum[r] = 0.f;
    }
    half_t* myP = Pt[wave];
#pragma unroll
    for (int j = 0; j < 4; j++) {
#pragma unroll
      for (int r = 0; r < 8; r++) {
        float p = expf(sc4[j][r] - mrow[r]);
        rsum[r] += p;
        myP[(r + 8 * half_id) * 64 + 16 * j + col] = (half_t)p;
      }
    }
#pragma unroll
    for (int r = 0; r < 8; r++) {
      float v = rsum[r];
#pragma unroll
      for (int off = 1; off < 16; off <<= 1) v += __shfl_xor(v, off, 32);
      lrow[r] = lrow[r] * scale[r] + v;
    }
#pragma unroll
    for (int f = 0; f < 4; f++)
#pragma unroll
      for (int r = 0; r < 8; r++) acc[f][r] *= scale[r];

    __syncthreads();  // Pt visible

    // --- ctx += P @ V --------------------------------------------------------
    v16h pa0 = load_fragA(myP, 64, 0, 0);
    v16h pa1 = load_fragA(myP, 64, 0, 32);
#pragma unroll
    for (int f = 0; f < 4; f++) {
      acc[f] = wmma_ff(pa0, load_fragB(Vt, 72, 16 * f, 0), acc[f]);
      acc[f] = wmma_ff(pa1, load_fragB(Vt, 72, 16 * f, 32), acc[f]);
    }
    __syncthreads();  // all reads of Kt/Vt done before next tile overwrite
  }

  // --- epilogue: ctx /= l ----------------------------------------------------
#pragma unroll
  for (int r = 0; r < 8; r++) lrow[r] = 1.0f / lrow[r];
#pragma unroll
  for (int f = 0; f < 4; f++)
#pragma unroll
    for (int r = 0; r < 8; r++) {
      int m = qrow + r + 8 * half_id;
      int n = 16 * f + col;
      Ctx[headoff + (size_t)m * HDIM + n] = (half_t)(acc[f][r] * lrow[r]);
    }
}

// ---------------------------------------------------------------------------
// In-place LayerNorm over rows of 768, one block per row.
__global__ __launch_bounds__(256) void ln_kernel(float* __restrict__ y,
                                                 const float* __restrict__ g,
                                                 const float* __restrict__ bb) {
  __shared__ float red[256];
  const int t = threadIdx.x;
  float* p = y + (size_t)blockIdx.x * HDIM;
  float v0 = p[t], v1 = p[t + 256], v2 = p[t + 512];
  red[t] = v0 + v1 + v2;
  __syncthreads();
  for (int o = 128; o > 0; o >>= 1) {
    if (t < o) red[t] += red[t + o];
    __syncthreads();
  }
  float mu = red[0] * (1.0f / 768.0f);
  __syncthreads();
  float d0 = v0 - mu, d1 = v1 - mu, d2 = v2 - mu;
  red[t] = d0 * d0 + d1 * d1 + d2 * d2;
  __syncthreads();
  for (int o = 128; o > 0; o >>= 1) {
    if (t < o) red[t] += red[t + o];
    __syncthreads();
  }
  float rstd = rsqrtf(red[0] * (1.0f / 768.0f) + 1e-12f);
  p[t]       = d0 * rstd * g[t]       + bb[t];
  p[t + 256] = d1 * rstd * g[t + 256] + bb[t + 256];
  p[t + 512] = d2 * rstd * g[t + 512] + bb[t + 512];
}

// ---------------------------------------------------------------------------
extern "C" void kernel_launch(void* const* d_in, const int* in_sizes, int n_in,
                              void* d_out, int out_size, void* d_ws, size_t ws_size,
                              hipStream_t stream) {
  (void)in_sizes; (void)n_in; (void)out_size; (void)ws_size;
  const float* hs   = (const float*)d_in[0];
  const float* mask = (const float*)d_in[1];
  const float* q_w  = (const float*)d_in[2];
  const float* q_b  = (const float*)d_in[3];
  const float* k_w  = (const float*)d_in[4];
  const float* k_b  = (const float*)d_in[5];
  const float* v_w  = (const float*)d_in[6];
  const float* v_b  = (const float*)d_in[7];
  const float* o_w  = (const float*)d_in[8];
  const float* o_b  = (const float*)d_in[9];
  const float* ln_g = (const float*)d_in[10];
  const float* ln_b = (const float*)d_in[11];
  float* outp = (float*)d_out;

  const size_t nHS = (size_t)MROWS * HDIM;  // 6,291,456
  const size_t nW  = (size_t)HDIM * HDIM;   // 589,824
  char* w = (char*)d_ws;
  half_t* hidden16 = (half_t*)w; w += nHS * 2;
  half_t* qw16 = (half_t*)w; w += nW * 2;
  half_t* kw16 = (half_t*)w; w += nW * 2;
  half_t* vw16 = (half_t*)w; w += nW * 2;
  half_t* ow16 = (half_t*)w; w += nW * 2;
  half_t* Q16  = (half_t*)w; w += nHS * 2;
  half_t* K16  = (half_t*)w; w += nHS * 2;
  half_t* V16  = (half_t*)w; w += nHS * 2;
  half_t* C16  = (half_t*)w; w += nHS * 2;

  // 1) f32 -> f16 staging
  convert_kernel<<<(unsigned)((nHS + 255) / 256), 256, 0, stream>>>(hs, hidden16, (int)nHS);
  convert_kernel<<<(unsigned)((nW + 255) / 256), 256, 0, stream>>>(q_w, qw16, (int)nW);
  convert_kernel<<<(unsigned)((nW + 255) / 256), 256, 0, stream>>>(k_w, kw16, (int)nW);
  convert_kernel<<<(unsigned)((nW + 255) / 256), 256, 0, stream>>>(v_w, vw16, (int)nW);
  convert_kernel<<<(unsigned)((nW + 255) / 256), 256, 0, stream>>>(o_w, ow16, (int)nW);

  // 2) QKV projections
  dim3 gg(HDIM / 128, MROWS / 128);
  gemm_qkv<<<gg, 256, 0, stream>>>(hidden16, qw16, q_b, Q16);
  gemm_qkv<<<gg, 256, 0, stream>>>(hidden16, kw16, k_b, K16);
  gemm_qkv<<<gg, 256, 0, stream>>>(hidden16, vw16, v_b, V16);

  // 3) flash attention
  dim3 ga(SD / 128, NHD, BD);
  attn_kernel<<<ga, 256, 0, stream>>>(Q16, K16, V16, mask, C16);

  // 4) output projection + bias + residual -> d_out (f32)
  gemm_oproj<<<gg, 256, 0, stream>>>(C16, ow16, o_b, hs, outp);

  // 5) in-place LayerNorm
  ln_kernel<<<MROWS, 256, 0, stream>>>(outp, ln_g, ln_b);
}
